// Head_54408645706053
// MI455X (gfx1250) — compile-verified, matrix operand order checked by reading, not run on hardware
//
#include <hip/hip_runtime.h>
#include <hip/hip_bf16.h>

typedef __attribute__((ext_vector_type(16))) __bf16 v16bf;
typedef __attribute__((ext_vector_type(8)))  __bf16 v8bf;
typedef __attribute__((ext_vector_type(8)))  float  v8f;
typedef __attribute__((ext_vector_type(16))) float  v16f;
typedef __attribute__((ext_vector_type(4)))  unsigned int v4u;
typedef __attribute__((ext_vector_type(8)))  int v8i;
typedef __attribute__((ext_vector_type(4)))  int v4i;

#define WMMA_BF16(A, B, C) \
    __builtin_amdgcn_wmma_f32_16x16x32_bf16(false, (A), false, (B), (short)0, (C), false, false)

__device__ __forceinline__ v16bf cat8(v8bf a, v8bf b) {
    return __builtin_shufflevector(a, b, 0,1,2,3,4,5,6,7,8,9,10,11,12,13,14,15);
}

// Fragment element i (of 16) -> K offset; lanes 0-15: K={0..7,16..23}, lanes 16-31: K={8..15,24..31}
// Runs of 8 are contiguous: [h*8, h*8+8) and [16+h*8, 16+h*8+8). Load each run as one b128.
__device__ __forceinline__ v16bf frag32(const __bf16* row, int hh) {
    v8bf lo = *(const v8bf*)(row + hh * 8);
    v8bf hi = *(const v8bf*)(row + 16 + hh * 8);
    return cat8(lo, hi);
}

static constexpr int S_LEN  = 4096;
static constexpr int DMODEL = 768;
static constexpr int DK     = 64;
static constexpr int ROWS   = 4 * 4096;   // B*S
static constexpr int BQ     = 64;
static constexpr int BK     = 64;

// ---------------------------------------------------------------------------
// TDM: 2D tile load (64 x 64 bf16) global -> LDS, per CDNA5 ISA D# layout.
// 6-arg builtin form: (v4u, v8i, v4i, v4i, v8i, i32 cpol).
// ---------------------------------------------------------------------------
#if __has_builtin(__builtin_amdgcn_tensor_load_to_lds)
#define HAVE_TDM 1
__device__ __forceinline__ unsigned lds_off(const void* p) {
#if defined(__HIP_DEVICE_COMPILE__)
    return (unsigned)(unsigned long long)(__attribute__((address_space(3))) const void*)p;
#else
    return 0;
#endif
}
__device__ __forceinline__ void tdm_load_2d(const void* g, unsigned lds,
                                            unsigned tensor_d0, unsigned tensor_d1,
                                            unsigned stride0) {
    unsigned long long ga = (unsigned long long)g;
    v4u g0 = { 1u,                                   // count=1 (valid descriptor)
               lds,                                  // lds_addr (bytes)
               (unsigned)ga,
               (unsigned)(ga >> 32) | (2u << 30) };  // global_addr hi | type=2
    v8i g1 = { (int)(1u << 16),                                          // data_size=2B
               (int)((tensor_d0 & 0xFFFFu) << 16),                       // tensor_dim0 lo
               (int)(((tensor_d0 >> 16) & 0xFFFFu) |
                     ((tensor_d1 & 0xFFFFu) << 16)),                     // dim0 hi | dim1 lo
               (int)(((tensor_d1 >> 16) & 0xFFFFu) | (64u << 16)),       // dim1 hi | tile_dim0=64
               (int)64,                                                  // tile_dim1=64
               (int)stride0,                                             // tensor_dim0_stride lo
               0, 0 };
    v4i z4 = {0, 0, 0, 0};
    v8i z8 = {0, 0, 0, 0, 0, 0, 0, 0};
    __builtin_amdgcn_tensor_load_to_lds(g0, g1, z4, z4, z8, 0);
}
#else
#define HAVE_TDM 0
#endif

// ---------------------------------------------------------------------------
// One-shot: W[768,64] f32 -> Wt[64,768] bf16 (transposed, pre-converted).
// grid = (192, 3), block = 256.
// ---------------------------------------------------------------------------
__global__ __launch_bounds__(256)
void wtrans_kernel(const float* __restrict__ WQ, const float* __restrict__ WK,
                   const float* __restrict__ WV, __bf16* __restrict__ WtQ,
                   __bf16* __restrict__ WtK, __bf16* __restrict__ WtV) {
    const float* W  = (blockIdx.y == 0) ? WQ  : (blockIdx.y == 1) ? WK  : WV;
    __bf16*      Wt = (blockIdx.y == 0) ? WtQ : (blockIdx.y == 1) ? WtK : WtV;
    const int idx = blockIdx.x * 256 + threadIdx.x;   // 0 .. 49151
    const int k = idx >> 6, n = idx & 63;
    Wt[(size_t)n * DMODEL + k] = (__bf16)W[idx];
}

// ---------------------------------------------------------------------------
// Projection: out = bf16( x[16384,768] @ W[768,64] ); W given transposed bf16.
// No LDS: A frags from x (4x global b128 + packed cvt), B frags from Wt (2x b128).
// TRANSPOSED=1 stores per-batch [dim][seq] (for V, so attention TDM can tile it).
// ---------------------------------------------------------------------------
template <int TRANSPOSED>
__global__ __launch_bounds__(128)
void proj_kernel(const float* __restrict__ x, const __bf16* __restrict__ Wt,
                 __bf16* __restrict__ outb) {
    const int tid  = threadIdx.x;
    const int wave = tid >> 5;
    const int lane = tid & 31;
    const int cn   = lane & 15;
    const int hh   = lane >> 4;
    const int row0 = blockIdx.x * 64 + wave * 16;

    v8f acc[4] = {};

    for (int kb = 0; kb < DMODEL; kb += 32) {
        const float* xr = x + (size_t)(row0 + cn) * DMODEL + kb;
        v8f fa = *(const v8f*)(xr + hh * 8);
        v8f fb = *(const v8f*)(xr + 16 + hh * 8);
        v16f xf = __builtin_shufflevector(fa, fb, 0,1,2,3,4,5,6,7,8,9,10,11,12,13,14,15);
        v16bf a = __builtin_convertvector(xf, v16bf);

        #pragma unroll
        for (int nt = 0; nt < 4; ++nt) {
            v16bf b = frag32(Wt + (size_t)(nt * 16 + cn) * DMODEL + kb, hh);
            acc[nt] = WMMA_BF16(a, b, acc[nt]);
        }
    }

    #pragma unroll
    for (int nt = 0; nt < 4; ++nt) {
        v8bf ob = __builtin_convertvector(acc[nt], v8bf);
        #pragma unroll
        for (int r = 0; r < 8; ++r) {
            const int row = row0 + r + 8 * hh;
            const int dim = nt * 16 + cn;
            if (TRANSPOSED) {
                const int b = row >> 12, s = row & (S_LEN - 1);
                outb[((size_t)b * DK + dim) * S_LEN + s] = ob[r];
            } else {
                outb[(size_t)row * DK + dim] = ob[r];
            }
        }
    }
}

// ---------------------------------------------------------------------------
// FlashAttention forward (causal). grid = (S/BQ, B), block = 128 (4 waves).
// K/V tiles staged by TDM (double-buffered); all fragments are b128 loads.
// ---------------------------------------------------------------------------
__global__ __launch_bounds__(128)
void attn_kernel(const __bf16* __restrict__ Qb, const __bf16* __restrict__ Kb,
                 const __bf16* __restrict__ Vtg, float* __restrict__ out) {
    __shared__ __align__(16) __bf16 Ks2[2][BK][DK];   // [key][dim]
    __shared__ __align__(16) __bf16 Vt2[2][DK][BK];   // [dim][key]
    __shared__ __align__(16) __bf16 Ps[4][16][BK];    // per-wave P slab [qrow][key]

    const int b    = blockIdx.y;
    const int qb   = blockIdx.x;
    const int tid  = threadIdx.x;
    const int wave = tid >> 5;
    const int lane = tid & 31;
    const int cn   = lane & 15;
    const int hh   = lane >> 4;
    const int q0   = qb * BQ + wave * 16;

    const __bf16* Qm = Qb  + (size_t)b * S_LEN * DK;
    const __bf16* Km = Kb  + (size_t)b * S_LEN * DK;
    const __bf16* Vg = Vtg + (size_t)b * DK * S_LEN;  // [dim][seq]

    const __bf16* qr = Qm + (size_t)(q0 + cn) * DK;
    v16bf aq0 = frag32(qr, hh);
    v16bf aq1 = frag32(qr + 32, hh);

    float m_i[8], l_i[8];
    v8f acc[4] = {};
    #pragma unroll
    for (int r = 0; r < 8; ++r) { m_i[r] = -1e30f; l_i[r] = 0.0f; }

#if HAVE_TDM
    if (wave == 0) {
        tdm_load_2d(Km, lds_off(&Ks2[0][0][0]), DK, S_LEN, DK);
        tdm_load_2d(Vg, lds_off(&Vt2[0][0][0]), S_LEN, DK, S_LEN);
    }
#endif

    for (int j = 0; j <= qb; ++j) {
        const int buf = j & 1;
#if HAVE_TDM
        if (wave == 0) {
            if (j < qb) {
                tdm_load_2d(Km + (size_t)(j + 1) * BK * DK, lds_off(&Ks2[buf ^ 1][0][0]),
                            DK, S_LEN, DK);
                tdm_load_2d(Vg + (size_t)(j + 1) * BK, lds_off(&Vt2[buf ^ 1][0][0]),
                            S_LEN, DK, S_LEN);
                __builtin_amdgcn_s_wait_tensorcnt(2);   // current pair done, next in flight
            } else {
                __builtin_amdgcn_s_wait_tensorcnt(0);
            }
        }
        __syncthreads();
#else
        __syncthreads();
        {
            const v8bf* ksrc = (const v8bf*)(Km + (size_t)j * BK * DK);
            v8bf* kdst = (v8bf*)&Ks2[buf][0][0];
            for (int i = tid; i < BK * DK / 8; i += 128) kdst[i] = ksrc[i];
            for (int i = tid; i < DK * BK / 8; i += 128) {
                const int d = i >> 3, kk = (i & 7) * 8;
                *(v8bf*)&Vt2[buf][d][kk] = *(const v8bf*)(Vg + (size_t)d * S_LEN + j * BK + kk);
            }
        }
        __syncthreads();
#endif

        // ---- scores: four 16x16 f32 tiles, B frags = 2x ds_load_b128 each ----
        v8f s[4];
        #pragma unroll
        for (int nt = 0; nt < 4; ++nt) {
            const __bf16* kr = &Ks2[buf][nt * 16 + cn][0];
            v8f c = {};
            c = WMMA_BF16(aq0, frag32(kr, hh), c);
            c = WMMA_BF16(aq1, frag32(kr + 32, hh), c);
            s[nt] = c;
        }

        const bool diag = (j == qb);
        #pragma unroll
        for (int nt = 0; nt < 4; ++nt)
            #pragma unroll
            for (int r = 0; r < 8; ++r) {
                float v = s[nt][r] * 0.125f;   // 1/sqrt(64)
                if (diag) {
                    const int qg = q0 + r + 8 * hh;
                    const int kg = j * BK + nt * 16 + cn;
                    if (kg > qg) v = -1e30f;
                }
                s[nt][r] = v;
            }

        // ---- online softmax (reductions stay within each 16-lane half) ----
        float mloc[8];
        #pragma unroll
        for (int r = 0; r < 8; ++r)
            mloc[r] = fmaxf(fmaxf(s[0][r], s[1][r]), fmaxf(s[2][r], s[3][r]));
        #pragma unroll
        for (int off = 1; off < 16; off <<= 1)
            #pragma unroll
            for (int r = 0; r < 8; ++r)
                mloc[r] = fmaxf(mloc[r], __shfl_xor(mloc[r], off, 32));

        #pragma unroll
        for (int r = 0; r < 8; ++r) {
            const float mnew = fmaxf(m_i[r], mloc[r]);
            const float corr = __expf(m_i[r] - mnew);
            m_i[r] = mnew;
            l_i[r] *= corr;
            #pragma unroll
            for (int nt = 0; nt < 4; ++nt) acc[nt][r] *= corr;
        }

        float rsum[8] = {};
        #pragma unroll
        for (int nt = 0; nt < 4; ++nt)
            #pragma unroll
            for (int r = 0; r < 8; ++r) {
                const float p = __expf(s[nt][r] - m_i[r]);
                s[nt][r] = p;
                rsum[r] += p;
            }
        #pragma unroll
        for (int off = 1; off < 16; off <<= 1)
            #pragma unroll
            for (int r = 0; r < 8; ++r)
                rsum[r] += __shfl_xor(rsum[r], off, 32);
        #pragma unroll
        for (int r = 0; r < 8; ++r) l_i[r] += rsum[r];

        // ---- P: C-layout -> LDS -> A-layout (same-wave LDS ordering) ----
        #pragma unroll
        for (int nt = 0; nt < 4; ++nt) {
            v8bf pv = __builtin_convertvector(s[nt], v8bf);
            #pragma unroll
            for (int r = 0; r < 8; ++r)
                Ps[wave][r + 8 * hh][nt * 16 + cn] = pv[r];
        }

        const __bf16* pr = &Ps[wave][cn][0];
        v16bf ap0 = frag32(pr, hh);
        v16bf ap1 = frag32(pr + 32, hh);

        #pragma unroll
        for (int nt = 0; nt < 4; ++nt) {
            const __bf16* vr = &Vt2[buf][nt * 16 + cn][0];
            acc[nt] = WMMA_BF16(ap0, frag32(vr, hh), acc[nt]);
            acc[nt] = WMMA_BF16(ap1, frag32(vr + 32, hh), acc[nt]);
        }

        __syncthreads();   // buffer reuse fence (prefetch of j+2 overwrites buf)
    }

    float inv[8];
    #pragma unroll
    for (int r = 0; r < 8; ++r) inv[r] = __builtin_amdgcn_rcpf(l_i[r]);
    #pragma unroll
    for (int nt = 0; nt < 4; ++nt)
        #pragma unroll
        for (int r = 0; r < 8; ++r) {
            const int row = r + 8 * hh;
            out[((size_t)b * S_LEN + q0 + row) * DK + nt * 16 + cn] = acc[nt][r] * inv[r];
        }
}

// ---------------------------------------------------------------------------
extern "C" void kernel_launch(void* const* d_in, const int* in_sizes, int n_in,
                              void* d_out, int out_size, void* d_ws, size_t ws_size,
                              hipStream_t stream) {
    const float* x  = (const float*)d_in[0];
    const float* WQ = (const float*)d_in[1];
    const float* WK = (const float*)d_in[2];
    const float* WV = (const float*)d_in[3];
    float* out = (float*)d_out;

    __bf16* Qb  = (__bf16*)d_ws;
    __bf16* Kb  = Qb + (size_t)ROWS * DK;
    __bf16* Vt  = Kb + (size_t)ROWS * DK;          // per-batch [dim][seq]
    __bf16* WtQ = Vt + (size_t)ROWS * DK;
    __bf16* WtK = WtQ + (size_t)DK * DMODEL;
    __bf16* WtV = WtK + (size_t)DK * DMODEL;

    wtrans_kernel<<<dim3(DMODEL * DK / 256, 3), dim3(256), 0, stream>>>(
        WQ, WK, WV, WtQ, WtK, WtV);

    dim3 pblk(128), pgrd(ROWS / 64);
    proj_kernel<0><<<pgrd, pblk, 0, stream>>>(x, WtQ, Qb);
    proj_kernel<0><<<pgrd, pblk, 0, stream>>>(x, WtK, Kb);
    proj_kernel<1><<<pgrd, pblk, 0, stream>>>(x, WtV, Vt);

    dim3 ablk(128), agrd(S_LEN / BQ, 4);
    attn_kernel<<<agrd, ablk, 0, stream>>>(Qb, Kb, Vt, out);
}
